// DeformableTransformerEncoderLayer_40948218200416
// MI455X (gfx1250) — matrix operand order repaired and checked by reading, not compiled, not run on hardware
//
#include <hip/hip_runtime.h>
#include <hip/hip_bf16.h>
#include <math.h>

#define BATCH   2
#define LINQ    13294
#define MTOT    (BATCH * LINQ)
#define DMODEL  256
#define DATTN   64
#define HATTN   8
#define HDIM    8
#define NLVLS   4
#define NPTSS   4
#define DDYN    192
#define HDYN    6
#define KWIN    7
#define PADLFT  3
#define DFFN    1024
#define EPSLN   1e-5f

typedef __attribute__((ext_vector_type(16))) __bf16          v16bf;
typedef __attribute__((ext_vector_type(8)))  float           v8f;
typedef __attribute__((ext_vector_type(4)))  float           f32x4;
typedef __attribute__((ext_vector_type(8)))  unsigned short  u16x8;   // 16B

union FragBF { v16bf v; u16x8 h[2]; unsigned short u[16]; };

__device__ __forceinline__ unsigned short f2bf(float f) {
    unsigned int x = __float_as_uint(f);
    unsigned int r = x + 0x7FFFu + ((x >> 16) & 1u);   // round-to-nearest-even
    return (unsigned short)(r >> 16);
}

// ---------------------------------------------------------------------------
// Tiled GEMM: Out[M,N](ldo) = act(A[M,K](lda) @ W[K,N] + bias), W row-major KxN.
// N, K compile-time -> immediate-offset staging loads, no per-iter addr math.
// Block = 256 threads = 8 waves; block tile 128(M) x 64(N), K staged by 32.
// Each wave owns a 16-row stripe, computes 16x64 via 4 WMMAs per K-step.
// A fragments: global b128 loads + f32->bf16 in regs (A is wave-private).
// B tile: staged transposed in LDS, fragment-contiguous -> ds_load_b128 pairs,
// all 8 fragment loads issued before the 4 WMMAs so waits can be partial.
// ---------------------------------------------------------------------------
template<int RELU, int N, int K>
__global__ void gemm_bf16_wmma(const float* __restrict__ A, int lda,
                               const float* __restrict__ W,
                               const float* __restrict__ bias,
                               float* __restrict__ Out, int ldo, int M)
{
    __shared__ __align__(16) unsigned short Bt[64][48];  // [col][k], 96B rows

    const int tid  = threadIdx.x;
    const int lane = tid & 31;
    const int wid  = tid >> 5;               // 0..7 -> wave's 16-row stripe
    const int m0   = blockIdx.y * 128;
    const int n0   = blockIdx.x * 64;

    v8f c0 = {}, c1 = {}, c2 = {}, c3 = {};

    // A fragment geometry (16x32 bf16 ISA layout)
    const int  arow = m0 + wid * 16 + (lane & 15);
    const bool aok  = arow < M;
    const float* aptr = A + (size_t)arow * lda;
    const int  ko   = (lane < 16) ? 0 : 8;

    // B staging geometry: thread -> (col c, k-rowgroup rg of 8 rows)
    const int c  = tid & 63;
    const int rg = tid >> 6;                 // 0..3
    const float* wp = W + (size_t)(rg * 8) * N + n0 + c;

    // B fragment geometry
    const int kb = (lane < 16) ? 0 : 8;
    const int cl = lane & 15;

    for (int k0 = 0; k0 < K; k0 += 32, wp += 32 * N) {
        // ---- stage B into regs: 8 loads, row offsets are immediates ----
        u16x8 pk;
        #pragma unroll
        for (int i = 0; i < 8; ++i)
            pk[i] = f2bf(wp[i * N]);
        if (k0 + 32 < K)
            __builtin_prefetch(wp + 32 * N, 0, 0);   // global_prefetch_b8

        // ---- A fragment from global, convert in regs ----
        FragBF a;
        if (aok) {
            f32x4 f0 = *(const f32x4*)(aptr + k0 + ko);
            f32x4 f1 = *(const f32x4*)(aptr + k0 + ko + 4);
            f32x4 f2 = *(const f32x4*)(aptr + k0 + ko + 16);
            f32x4 f3 = *(const f32x4*)(aptr + k0 + ko + 20);
            #pragma unroll
            for (int j = 0; j < 4; ++j) {
                a.u[j]      = f2bf(f0[j]);
                a.u[4 + j]  = f2bf(f1[j]);
                a.u[8 + j]  = f2bf(f2[j]);
                a.u[12 + j] = f2bf(f3[j]);
            }
        } else {
            #pragma unroll
            for (int j = 0; j < 16; ++j) a.u[j] = 0;
        }

        __syncthreads();                      // prior reads of Bt done
        *(u16x8*)&Bt[c][rg * 8] = pk;         // one ds_store_b128 per thread
        __syncthreads();                      // Bt ready

        // ---- load ALL 4 B fragments first (8x ds_load_b128 clause) ----
        FragBF b0, b1, b2, b3;
        b0.h[0] = *(const u16x8*)&Bt[cl     ][kb];
        b0.h[1] = *(const u16x8*)&Bt[cl     ][kb + 16];
        b1.h[0] = *(const u16x8*)&Bt[cl + 16][kb];
        b1.h[1] = *(const u16x8*)&Bt[cl + 16][kb + 16];
        b2.h[0] = *(const u16x8*)&Bt[cl + 32][kb];
        b2.h[1] = *(const u16x8*)&Bt[cl + 32][kb + 16];
        b3.h[0] = *(const u16x8*)&Bt[cl + 48][kb];
        b3.h[1] = *(const u16x8*)&Bt[cl + 48][kb + 16];

        c0 = __builtin_amdgcn_wmma_f32_16x16x32_bf16(false, a.v, false, b0.v, (short)0, c0, false, false);
        c1 = __builtin_amdgcn_wmma_f32_16x16x32_bf16(false, a.v, false, b1.v, (short)0, c1, false, false);
        c2 = __builtin_amdgcn_wmma_f32_16x16x32_bf16(false, a.v, false, b2.v, (short)0, c2, false, false);
        c3 = __builtin_amdgcn_wmma_f32_16x16x32_bf16(false, a.v, false, b3.v, (short)0, c3, false, false);
    }

    // ---- epilogue: C/D layout rows = +0 (lanes<16) / +8 (lanes>=16) ----
    const int rowb = m0 + wid * 16 + ((lane < 16) ? 0 : 8);
    #pragma unroll
    for (int g = 0; g < 4; ++g) {
        const int col = n0 + g * 16 + cl;
        const float bb = bias ? bias[col] : 0.0f;
        const v8f& cc = (g == 0) ? c0 : (g == 1) ? c1 : (g == 2) ? c2 : c3;
        #pragma unroll
        for (int rr = 0; rr < 8; ++rr) {
            const int grow = rowb + rr;
            if (grow < M) {
                float v = cc[rr] + bb;
                if (RELU) v = fmaxf(v, 0.0f);
                Out[(size_t)grow * ldo + col] = v;
            }
        }
    }
}

// ---------------------------------------------------------------------------
__global__ void add_q_kernel(const float* __restrict__ src,
                             const float* __restrict__ pos,
                             float* __restrict__ q)
{
    size_t i = (size_t)blockIdx.x * blockDim.x + threadIdx.x;
    if (i < (size_t)MTOT * DMODEL) q[i] = src[i] + pos[i];
}

__global__ void softmax16_kernel(float* __restrict__ aw)
{
    int tid = blockIdx.x * blockDim.x + threadIdx.x;
    if (tid >= MTOT * HATTN) return;
    float* p = aw + (size_t)tid * 16;
    float mx = p[0];
    #pragma unroll
    for (int i = 1; i < 16; ++i) mx = fmaxf(mx, p[i]);
    float s = 0.0f; float e[16];
    #pragma unroll
    for (int i = 0; i < 16; ++i) { e[i] = __expf(p[i] - mx); s += e[i]; }
    float inv = 1.0f / s;
    #pragma unroll
    for (int i = 0; i < 16; ++i) p[i] = e[i] * inv;
}

__global__ void dynw_logits_kernel(const float* __restrict__ xd,
                                   const float* __restrict__ Ww,
                                   const float* __restrict__ bw,
                                   float* __restrict__ wl)
{
    int tid = blockIdx.x * blockDim.x + threadIdx.x;
    if (tid >= MTOT * (HDYN * KWIN)) return;
    const int m = tid / (HDYN * KWIN);
    const int j = tid - m * (HDYN * KWIN);
    const float* xp = xd + (size_t)m * DDYN;
    float acc = bw[j];
    #pragma unroll 8
    for (int k = 0; k < DDYN; ++k) acc += xp[k] * Ww[k * (HDYN * KWIN) + j];
    wl[(size_t)m * (HDYN * KWIN) + j] = acc;
}

__global__ void softmax7_kernel(float* __restrict__ wl)
{
    int tid = blockIdx.x * blockDim.x + threadIdx.x;
    if (tid >= MTOT * HDYN) return;
    float* p = wl + (size_t)tid * KWIN;
    float mx = p[0];
    #pragma unroll
    for (int i = 1; i < KWIN; ++i) mx = fmaxf(mx, p[i]);
    float s = 0.0f; float e[KWIN];
    #pragma unroll
    for (int i = 0; i < KWIN; ++i) { e[i] = __expf(p[i] - mx); s += e[i]; }
    float inv = 1.0f / s;
    #pragma unroll
    for (int i = 0; i < KWIN; ++i) p[i] = e[i] * inv;
}

// ---------------------------------------------------------------------------
__global__ void sample_kernel(const float* __restrict__ value,
                              const float* __restrict__ off,
                              const float* __restrict__ aw,
                              const float* __restrict__ ref,
                              float* __restrict__ samp)
{
    int tid = blockIdx.x * blockDim.x + threadIdx.x;
    if (tid >= MTOT * HATTN) return;
    const int m = tid >> 3;
    const int h = tid & 7;
    const int b = m / LINQ;

    const int Hl[NLVLS] = {100, 50, 25, 13};
    const int Wl[NLVLS] = {100, 50, 25, 13};
    const int st[NLVLS] = {0, 10000, 12500, 13125};

    f32x4 acc0 = {}, acc1 = {};
    const float* refp = ref + (size_t)m * NLVLS * 2;

    for (int lvl = 0; lvl < NLVLS; ++lvl) {
        const float fw = (float)Wl[lvl], fh = (float)Hl[lvl];
        const float rx = refp[lvl * 2], ry = refp[lvl * 2 + 1];
        #pragma unroll
        for (int p = 0; p < NPTSS; ++p) {
            const int oi = ((h * NLVLS + lvl) * NPTSS + p) * 2;
            const float ox = off[(size_t)m * 512 + oi];
            const float oy = off[(size_t)m * 512 + oi + 1];
            const float x = (rx + ox / fw) * fw - 0.5f;
            const float y = (ry + oy / fh) * fh - 0.5f;
            const float x0 = floorf(x), y0 = floorf(y);
            const float a = aw[(size_t)m * 128 + h * 16 + lvl * NPTSS + p];
            #pragma unroll
            for (int dx = 0; dx < 2; ++dx)
            #pragma unroll
            for (int dy = 0; dy < 2; ++dy) {
                const float xi = x0 + dx, yi = y0 + dy;
                if (xi >= 0.0f && xi < fw && yi >= 0.0f && yi < fh) {
                    const float w = (1.0f - fabsf(x - xi)) * (1.0f - fabsf(y - yi)) * a;
                    const int idx = (int)yi * Wl[lvl] + (int)xi;
                    const float* vp = value +
                        ((size_t)(b * LINQ + st[lvl] + idx) * DATTN + h * HDIM);
                    f32x4 v0 = *(const f32x4*)vp;
                    f32x4 v1 = *(const f32x4*)(vp + 4);
                    acc0 += w * v0;
                    acc1 += w * v1;
                }
            }
        }
    }
    float* op = samp + (size_t)m * DATTN + h * HDIM;
    *(f32x4*)op       = acc0;
    *(f32x4*)(op + 4) = acc1;
}

__global__ void dynconv_kernel(const float* __restrict__ xd,
                               const float* __restrict__ wl,
                               float* __restrict__ dyn)
{
    int tid = blockIdx.x * blockDim.x + threadIdx.x;
    if (tid >= MTOT * DDYN) return;
    const int m = tid / DDYN;
    const int cc = tid - m * DDYN;
    const int h = cc >> 5;
    const int b = m / LINQ;
    const int t = m - b * LINQ;
    const float* wp = wl + (size_t)m * (HDYN * KWIN) + h * KWIN;
    float acc = 0.0f;
    #pragma unroll
    for (int k = 0; k < KWIN; ++k) {
        const int tt = t + k - PADLFT;
        if (tt >= 0 && tt < LINQ)
            acc += xd[(size_t)(b * LINQ + tt) * DDYN + cc] * wp[k];
    }
    dyn[(size_t)m * DDYN + cc] = acc;
}

__global__ void add_ln_kernel(const float* __restrict__ X,
                              const float* __restrict__ Y,
                              const float* __restrict__ g,
                              const float* __restrict__ be,
                              float* __restrict__ Out)
{
    __shared__ float r1[DMODEL];
    __shared__ float r2[DMODEL];
    const int row = blockIdx.x;
    const int t = threadIdx.x;
    const float v = X[(size_t)row * DMODEL + t] + Y[(size_t)row * DMODEL + t];
    r1[t] = v; r2[t] = v * v;
    __syncthreads();
    for (int s = DMODEL / 2; s > 0; s >>= 1) {
        if (t < s) { r1[t] += r1[t + s]; r2[t] += r2[t + s]; }
        __syncthreads();
    }
    const float mean = r1[0] * (1.0f / DMODEL);
    const float var  = r2[0] * (1.0f / DMODEL) - mean * mean;
    const float inv  = rsqrtf(var + EPSLN);
    Out[(size_t)row * DMODEL + t] = (v - mean) * inv * g[t] + be[t];
}

// ---------------------------------------------------------------------------
extern "C" void kernel_launch(void* const* d_in, const int* in_sizes, int n_in,
                              void* d_out, int out_size, void* d_ws, size_t ws_size,
                              hipStream_t stream)
{
    const float* src  = (const float*)d_in[0];
    const float* pos  = (const float*)d_in[1];
    const float* ref  = (const float*)d_in[2];
    const float* Wv   = (const float*)d_in[5];
    const float* bv   = (const float*)d_in[6];
    const float* Woff = (const float*)d_in[7];
    const float* boff = (const float*)d_in[8];
    const float* Watt = (const float*)d_in[9];
    const float* batt = (const float*)d_in[10];
    const float* Wout = (const float*)d_in[11];
    const float* bout = (const float*)d_in[12];
    const float* W1   = (const float*)d_in[13];
    const float* b1   = (const float*)d_in[14];
    const float* Ww   = (const float*)d_in[15];
    const float* bw   = (const float*)d_in[16];
    const float* W2   = (const float*)d_in[17];
    const float* b2   = (const float*)d_in[18];
    const float* g1   = (const float*)d_in[19];
    const float* be1  = (const float*)d_in[20];
    const float* Wf1  = (const float*)d_in[21];
    const float* bf1  = (const float*)d_in[22];
    const float* Wf2  = (const float*)d_in[23];
    const float* bf2  = (const float*)d_in[24];
    const float* g2   = (const float*)d_in[25];
    const float* be2  = (const float*)d_in[26];
    float* out = (float*)d_out;

    float* q    = (float*)d_ws;                      // [M,256]
    float* val  = q    + (size_t)MTOT * 256;         // [M,64]
    float* off  = val  + (size_t)MTOT * 64;          // [M,512]
    float* aw   = off  + (size_t)MTOT * 512;         // [M,128]
    float* samp = aw   + (size_t)MTOT * 128;         // [M,64]
    float* xd   = samp + (size_t)MTOT * 64;          // [M,192]
    float* wl   = xd   + (size_t)MTOT * 192;         // [M,42]
    float* dyn  = wl   + (size_t)MTOT * 42;          // [M,192]
    float* src2 = dyn  + (size_t)MTOT * 192;         // [M,256]
    float* hbuf = src2 + (size_t)MTOT * 256;         // [M,256]
    float* ff1  = hbuf + (size_t)MTOT * 256;         // [M,1024]
    float* ffo  = ff1  + (size_t)MTOT * 1024;        // [M,256]

    const int MT = (MTOT + 127) / 128;   // 208 M-tiles
    const dim3 blk(256);

    add_q_kernel<<<((size_t)MTOT * 256 + 255) / 256, blk, 0, stream>>>(src, pos, q);

    gemm_bf16_wmma<0,  64,   64><<<dim3(1,  MT), blk, 0, stream>>>(src,       DMODEL, Wv,   bv,   val,  DATTN,  MTOT);
    gemm_bf16_wmma<0, 512,   64><<<dim3(8,  MT), blk, 0, stream>>>(q,         DMODEL, Woff, boff, off,  512,    MTOT);
    gemm_bf16_wmma<0, 128,   64><<<dim3(2,  MT), blk, 0, stream>>>(q,         DMODEL, Watt, batt, aw,   128,    MTOT);
    softmax16_kernel<<<(MTOT * HATTN + 255) / 256, blk, 0, stream>>>(aw);
    gemm_bf16_wmma<0, 192,  192><<<dim3(3,  MT), blk, 0, stream>>>(q + DATTN, DMODEL, W1,   b1,   xd,   DDYN,   MTOT);

    dynw_logits_kernel<<<(MTOT * HDYN * KWIN + 255) / 256, blk, 0, stream>>>(xd, Ww, bw, wl);
    softmax7_kernel<<<(MTOT * HDYN + 255) / 256, blk, 0, stream>>>(wl);

    sample_kernel<<<(MTOT * HATTN + 255) / 256, blk, 0, stream>>>(val, off, aw, ref, samp);

    gemm_bf16_wmma<0,  64,   64><<<dim3(1,  MT), blk, 0, stream>>>(samp, DATTN, Wout, bout, src2,         DMODEL, MTOT);
    dynconv_kernel<<<((size_t)MTOT * DDYN + 255) / 256, blk, 0, stream>>>(xd, wl, dyn);
    gemm_bf16_wmma<0, 192,  192><<<dim3(3,  MT), blk, 0, stream>>>(dyn,  DDYN,  W2,   b2,   src2 + DATTN, DMODEL, MTOT);

    add_ln_kernel<<<MTOT, dim3(DMODEL), 0, stream>>>(src, src2, g1, be1, hbuf);

    gemm_bf16_wmma<1, 1024, 256><<<dim3(16, MT), blk, 0, stream>>>(hbuf, DMODEL, Wf1, bf1, ff1, DFFN,   MTOT);
    gemm_bf16_wmma<0, 256, 1024><<<dim3(4,  MT), blk, 0, stream>>>(ff1,  DFFN,   Wf2, bf2, ffo, DMODEL, MTOT);

    add_ln_kernel<<<MTOT, dim3(DMODEL), 0, stream>>>(hbuf, ffo, g2, be2, out);
}
